// LISILoss_58506044506816
// MI455X (gfx1250) — matrix operand order repaired and checked by reading, not compiled, not run on hardware
//
#include <hip/hip_runtime.h>
#include <cstdint>

#define N_PTS 8192
#define DIM 64
#define KP1 91   // K+1 including self
#define KNN 90
#define NCAT 8
#define NTRIES 50
#define TOLV 1e-5f

typedef __attribute__((ext_vector_type(16))) __bf16 v16bf;
typedef __attribute__((ext_vector_type(8)))  float  v8f;

static __device__ __forceinline__ unsigned short f32_to_bf16_rne(float f) {
    union { float f; unsigned u; } x; x.f = f;
    unsigned u = x.u;
    unsigned r = u + 0x7FFFu + ((u >> 16) & 1u);
    return (unsigned short)(r >> 16);
}
static __device__ __forceinline__ float bf16_to_f32(unsigned short h) {
    union { unsigned u; float f; } x; x.u = ((unsigned)h) << 16; return x.f;
}

// CDNA5 async global->LDS copy (ASYNCcnt path).  Per-lane: 16B global -> 16B LDS.
static __device__ __forceinline__ void async_copy_b128(unsigned lds_addr, const void* gptr) {
    asm volatile("global_load_async_to_lds_b128 %0, %1, off"
                 :: "v"(lds_addr), "v"(gptr)
                 : "memory");
}
static __device__ __forceinline__ void wait_async_zero() {
    asm volatile("s_wait_asynccnt 0x0" ::: "memory");
}
// Generic pointer -> 32-bit LDS offset (LDS aperture: addr[31:0] is the LDS address)
static __device__ __forceinline__ unsigned lds_off(const void* p) {
    return (unsigned)(uintptr_t)p;
}

// ---------------------------------------------------------------------------
// Kernel 1: per-row squared norms + split fp32 -> bf16 hi/lo pair
// ---------------------------------------------------------------------------
__global__ __launch_bounds__(256)
void prep_kernel(const float* __restrict__ x, float* __restrict__ sq,
                 unsigned short* __restrict__ xhi, unsigned short* __restrict__ xlo) {
    int row = blockIdx.x * blockDim.x + threadIdx.x;
    if (row >= N_PTS) return;
    float acc = 0.f;
#pragma unroll
    for (int d = 0; d < DIM; ++d) {
        float v = x[row * DIM + d];
        acc += v * v;
        unsigned short hi = f32_to_bf16_rne(v);
        float rem = v - bf16_to_f32(hi);
        unsigned short lo = f32_to_bf16_rne(rem);
        xhi[row * DIM + d] = hi;
        xlo[row * DIM + d] = lo;
    }
    sq[row] = acc;
}

// ---------------------------------------------------------------------------
// Kernel 2: D2 = sq_i + sq_j - 2 * (x x^T) via split-bf16 WMMA
//   block = 256 threads = 8 waves; tile = 16 rows x 128 cols.
//   A/B hi+lo staged once per block via async global->LDS (36 KB of LDS),
//   fragments loaded from LDS as contiguous b128 chunks.  grid = (N/128, N/16)
// ---------------------------------------------------------------------------
union FragBF { unsigned u[8]; uint4 q[2]; v16bf v; };

__global__ __launch_bounds__(256)
void d2_wmma_kernel(const unsigned short* __restrict__ xhi,
                    const unsigned short* __restrict__ xlo,
                    const float* __restrict__ sq,
                    float* __restrict__ d2) {
    __shared__ __align__(16) unsigned short sAh[16 * DIM],  sAl[16 * DIM];
    __shared__ __align__(16) unsigned short sBh[128 * DIM], sBl[128 * DIM];

    const int tid   = threadIdx.x;
    const int lane  = tid & 31;
    const int wave  = tid >> 5;
    const int half  = lane >> 4;
    const int lm    = lane & 15;
    const int rowA0 = blockIdx.y * 16;
    const int colB0 = blockIdx.x * 128;

    // ---- stage A: 16x64 ushort (hi and lo) = 2x128 b128 chunks; 1 per thread
    {
        int ch = tid & 127;                  // chunk within hi (tid<128) or lo
        int r  = ch >> 3, dc = (ch & 7) * 8;
        const unsigned short* src = ((tid < 128) ? xhi : xlo) + (size_t)(rowA0 + r) * DIM + dc;
        unsigned short* dst = ((tid < 128) ? sAh : sAl) + r * DIM + dc;
        async_copy_b128(lds_off(dst), src);
    }
    // ---- stage B: 128x64 ushort (hi and lo) = 2x1024 b128 chunks; 8 per thread
#pragma unroll
    for (int c = 0; c < 4; ++c) {
        int ch = tid + c * 256;              // [0,1024)
        int r  = ch >> 3, dc = (ch & 7) * 8;
        async_copy_b128(lds_off(&sBh[r * DIM + dc]), xhi + (size_t)(colB0 + r) * DIM + dc);
        async_copy_b128(lds_off(&sBl[r * DIM + dc]), xlo + (size_t)(colB0 + r) * DIM + dc);
    }
    wait_async_zero();
    __syncthreads();

    v8f c = {};
#pragma unroll
    for (int kc = 0; kc < 2; ++kc) {         // K chunks of 32 over DIM=64
        FragBF ah, al, bh, bl;
        // A 16x32 bf16 fragment: lane holds K = {base..base+7} and {base+16..base+23}
        //   where base = kc*32 + (half?8:0); both runs are 16B-contiguous in LDS.
        const int aoff = lm * DIM + kc * 32 + (half ? 8 : 0);       // ushort units
        // B 32x16 bf16 fragment: same K layout as A (runs at base and base+16).
        const int boff = (wave * 16 + lm) * DIM + kc * 32 + (half ? 8 : 0);

        ah.q[0] = *(const uint4*)(sAh + aoff);
        ah.q[1] = *(const uint4*)(sAh + aoff + 16);
        al.q[0] = *(const uint4*)(sAl + aoff);
        al.q[1] = *(const uint4*)(sAl + aoff + 16);
        bh.q[0] = *(const uint4*)(sBh + boff);
        bh.q[1] = *(const uint4*)(sBh + boff + 16);
        bl.q[0] = *(const uint4*)(sBl + boff);
        bl.q[1] = *(const uint4*)(sBl + boff + 16);

        // dot ~= hi*hi + hi*lo + lo*hi  (lo*lo negligible, below fp32 ulp here)
        c = __builtin_amdgcn_wmma_f32_16x16x32_bf16(false, ah.v, false, bh.v,
                                                    (short)0, c, false, false);
        c = __builtin_amdgcn_wmma_f32_16x16x32_bf16(false, ah.v, false, bl.v,
                                                    (short)0, c, false, false);
        c = __builtin_amdgcn_wmma_f32_16x16x32_bf16(false, al.v, false, bh.v,
                                                    (short)0, c, false, false);
    }

    const int col = colB0 + wave * 16 + lm;
    const float sqc = sq[col];
#pragma unroll
    for (int v = 0; v < 8; ++v) {
        // C/D layout: VGPR v -> M = v (lanes 0-15) or v+8 (lanes 16-31); N = lane%16
        int row = rowA0 + v + (half ? 8 : 0);
        float val = sq[row] + sqc - 2.0f * c[v];
        val = fmaxf(val, 0.0f);
        d2[(long long)row * N_PTS + col] = val;
    }
}

// ---------------------------------------------------------------------------
// Kernel 3: exact top-(K+1) smallest per row via MSB radix-select.
//   Slot 0 gets the row minimum (self); remaining 90 slots get the rest of the
//   91 smallest in deterministic (index-ordered) compaction.  Order of slots
//   1..90 is irrelevant downstream (all reductions are symmetric).
//   one block (256 threads) per row; each thread owns a contiguous 32-chunk.
// ---------------------------------------------------------------------------
__global__ __launch_bounds__(256)
void topk_kernel(const float* __restrict__ d2,
                 float* __restrict__ topd, int* __restrict__ topi) {
    __shared__ float    sf[N_PTS];       // 32 KB row cache
    __shared__ unsigned hist[256];
    __shared__ unsigned cnt[256];
    __shared__ unsigned cnt2[256];
    __shared__ float    s_minv[256];
    __shared__ int      s_mini[256];
    __shared__ unsigned sh_prefix;
    __shared__ unsigned sh_krem;

    const int row = blockIdx.x;
    const int tid = threadIdx.x;
    const int i0  = tid * 32;            // this thread's contiguous chunk

    for (int i = tid; i < N_PTS; i += 256)
        sf[i] = d2[(long long)row * N_PTS + i];
    __syncthreads();

    // ---- argmin (self) -------------------------------------------------
    {
        float bv = __builtin_inff(); int bi = 0;
        for (int i = i0; i < i0 + 32; ++i) {
            float v = sf[i];
            if (v < bv || (v == bv && i < bi)) { bv = v; bi = i; }
        }
        s_minv[tid] = bv; s_mini[tid] = bi;
        __syncthreads();
        for (int st = 128; st > 0; st >>= 1) {
            if (tid < st) {
                float ov = s_minv[tid + st]; int oi = s_mini[tid + st];
                if (ov < s_minv[tid] || (ov == s_minv[tid] && oi < s_mini[tid])) {
                    s_minv[tid] = ov; s_mini[tid] = oi;
                }
            }
            __syncthreads();
        }
    }
    const float minv = s_minv[0];
    const int   mini = s_mini[0];
    if (tid == 0) { topd[row * KP1] = minv; topi[row * KP1] = mini; }

    // ---- radix select: V = bit pattern of the 91st smallest ------------
    // all values >= 0  =>  uint order == float order
    if (tid == 0) { sh_prefix = 0u; sh_krem = KP1; }
    __syncthreads();
    for (int shift = 24; shift >= 0; shift -= 8) {
        hist[tid] = 0u;
        __syncthreads();
        const unsigned prefix = sh_prefix;
        const unsigned krem   = sh_krem;
        const unsigned maskhi = (shift == 24) ? 0u : (0xFFFFFFFFu << (shift + 8));
        for (int i = i0; i < i0 + 32; ++i) {
            unsigned u = __float_as_uint(sf[i]);
            if ((u & maskhi) == prefix)
                atomicAdd(&hist[(u >> shift) & 255u], 1u);
        }
        __syncthreads();
        cnt[tid] = hist[tid];
        __syncthreads();
        for (int off = 1; off < 256; off <<= 1) {   // inclusive Hillis-Steele scan
            unsigned v = (tid >= off) ? cnt[tid - off] : 0u;
            __syncthreads();
            cnt[tid] += v;
            __syncthreads();
        }
        unsigned below = (tid == 0) ? 0u : cnt[tid - 1];
        if (cnt[tid] >= krem && below < krem) {     // unique winner digit
            sh_prefix = prefix | ((unsigned)tid << shift);
            sh_krem   = krem - below;
        }
        __syncthreads();
    }
    const unsigned V = sh_prefix;

    // ---- deterministic compaction --------------------------------------
    unsigned myLT = 0, myEQ = 0;
    for (int i = i0; i < i0 + 32; ++i) {
        unsigned u = __float_as_uint(sf[i]);
        if (i == mini) continue;
        if (u < V) myLT++;
        else if (u == V) myEQ++;
    }
    cnt[tid] = myLT; cnt2[tid] = myEQ;
    __syncthreads();
    for (int off = 1; off < 256; off <<= 1) {       // inclusive scans (both)
        unsigned v1 = (tid >= off) ? cnt[tid - off]  : 0u;
        unsigned v2 = (tid >= off) ? cnt2[tid - off] : 0u;
        __syncthreads();
        cnt[tid] += v1; cnt2[tid] += v2;
        __syncthreads();
    }
    const unsigned totLT = cnt[255];
    unsigned baseLT = 1u + ((tid == 0) ? 0u : cnt[tid - 1]);
    unsigned baseEQ = 1u + totLT + ((tid == 0) ? 0u : cnt2[tid - 1]);
    for (int i = i0; i < i0 + 32; ++i) {
        unsigned u = __float_as_uint(sf[i]);
        if (i == mini) continue;
        if (u < V) {
            unsigned p = baseLT++;
            if (p < KP1) { topd[row * KP1 + p] = sf[i]; topi[row * KP1 + p] = i; }
        } else if (u == V) {
            unsigned p = baseEQ++;                  // ties taken in index order
            if (p < KP1) { topd[row * KP1 + p] = sf[i]; topi[row * KP1 + p] = i; }
        }
    }
}

// ---------------------------------------------------------------------------
// Kernel 4: beta bisection (50 iters) + Simpson index; one wave32 per row
// ---------------------------------------------------------------------------
__global__ __launch_bounds__(256)
void lisi_kernel(const float* __restrict__ topd, const int* __restrict__ topi,
                 const int* __restrict__ batch_id, float* __restrict__ out) {
    const int lane = threadIdx.x & 31;
    const int row  = blockIdx.x * 8 + (threadIdx.x >> 5);
    if (row >= N_PTS) return;

    float d[3]; bool valid[3]; int lbl[3];
#pragma unroll
    for (int t = 0; t < 3; ++t) {
        int j = lane + 32 * t;
        valid[t] = (j < KNN);
        if (valid[t]) {
            float v = topd[row * KP1 + 1 + j];          // drop self (slot 0)
            d[t]   = sqrtf(fmaxf(v, 0.0f));
            lbl[t] = batch_id[topi[row * KP1 + 1 + j]];
        } else { d[t] = 0.f; lbl[t] = -1; }
    }

    const float logU = logf(30.0f);
    float beta = 1.f;
    float bmin = -__builtin_inff(), bmax = __builtin_inff();
    float P[3], H, Hdiff;

    auto hbeta = [&](float b, float* Pn, float& Hout) {
        float p[3]; float ps = 0.f, dp = 0.f;
#pragma unroll
        for (int t = 0; t < 3; ++t) {
            p[t] = valid[t] ? expf(-d[t] * b) : 0.f;
            ps += p[t];
            dp += d[t] * p[t];
        }
#pragma unroll
        for (int m = 16; m > 0; m >>= 1) {
            ps += __shfl_xor(ps, m, 32);
            dp += __shfl_xor(dp, m, 32);
        }
        bool safe = ps > 0.f;
        float psafe = safe ? ps : 1.f;
        Hout = safe ? (logf(psafe) + b * dp / psafe) : 0.f;
#pragma unroll
        for (int t = 0; t < 3; ++t) Pn[t] = safe ? p[t] / psafe : 0.f;
    };

    hbeta(beta, P, H);
    Hdiff = H - logU;

    for (int it = 0; it < NTRIES; ++it) {
        bool done = fabsf(Hdiff) < TOLV;
        bool pos  = Hdiff > 0.f;
        float nbmin = pos ? beta : bmin;
        float nbmax = pos ? bmax : beta;
        float b_up = (bmax <  __builtin_inff()) ? (beta + bmax) * 0.5f : beta * 2.f;
        float b_dn = (bmin > -__builtin_inff()) ? (beta + bmin) * 0.5f : beta * 0.5f;
        float nbeta = pos ? b_up : b_dn;
        beta = done ? beta : nbeta;
        bmin = done ? bmin : nbmin;
        bmax = done ? bmax : nbmax;
        float Pn[3], Hn;
        hbeta(beta, Pn, Hn);
        H = done ? H : Hn;
#pragma unroll
        for (int t = 0; t < 3; ++t) P[t] = done ? P[t] : Pn[t];
        Hdiff = H - logU;
    }

    float acc[NCAT];
#pragma unroll
    for (int c = 0; c < NCAT; ++c) acc[c] = 0.f;
#pragma unroll
    for (int t = 0; t < 3; ++t) {
#pragma unroll
        for (int c = 0; c < NCAT; ++c)
            acc[c] += (lbl[t] == c) ? P[t] : 0.f;
    }
    float simpson = 0.f;
#pragma unroll
    for (int c = 0; c < NCAT; ++c) {
        float s = acc[c];
#pragma unroll
        for (int m = 16; m > 0; m >>= 1) s += __shfl_xor(s, m, 32);
        simpson += s * s;
    }
    if (H == 0.f) simpson += -1.0f;
    if (lane == 0) out[row] = 1.0f / simpson;
}

// ---------------------------------------------------------------------------
extern "C" void kernel_launch(void* const* d_in, const int* in_sizes, int n_in,
                              void* d_out, int out_size, void* d_ws, size_t ws_size,
                              hipStream_t stream) {
    const float* x        = (const float*)d_in[0];
    const int*   batch_id = (const int*)d_in[1];
    float*       out      = (float*)d_out;

    char* ws = (char*)d_ws;
    size_t off = 0;
    float*          sq   = (float*)(ws + off);          off += (size_t)N_PTS * 4;
    unsigned short* xhi  = (unsigned short*)(ws + off); off += (size_t)N_PTS * DIM * 2;
    unsigned short* xlo  = (unsigned short*)(ws + off); off += (size_t)N_PTS * DIM * 2;
    float*          topd = (float*)(ws + off);          off += (size_t)N_PTS * KP1 * 4;
    int*            topi = (int*)(ws + off);            off += (size_t)N_PTS * KP1 * 4;
    off = (off + 255) & ~(size_t)255;
    float*          d2   = (float*)(ws + off);          // N*N*4 = 256 MB

    prep_kernel<<<N_PTS / 256, 256, 0, stream>>>(x, sq, xhi, xlo);

    dim3 g(N_PTS / 128, N_PTS / 16);
    d2_wmma_kernel<<<g, 256, 0, stream>>>(xhi, xlo, sq, d2);

    topk_kernel<<<N_PTS, 256, 0, stream>>>(d2, topd, topi);

    lisi_kernel<<<N_PTS / 8, 256, 0, stream>>>(topd, topi, batch_id, out);
}
